// SimpleDecoder_22368189677667
// MI455X (gfx1250) — matrix operand order repaired and checked by reading, not compiled, not run on hardware
//
#include <hip/hip_runtime.h>

typedef __bf16 bfT;
typedef __attribute__((ext_vector_type(16))) __bf16 v16bf;
typedef __attribute__((ext_vector_type(8)))  float  v8f;
typedef __attribute__((ext_vector_type(4)))  float  f32x4;
typedef __attribute__((ext_vector_type(4)))  unsigned int u32x4;
typedef __attribute__((ext_vector_type(8)))  int    i32x8;
typedef __attribute__((ext_vector_type(4)))  int    i32x4;

#define HIDDEN 1024
#define FFDIM  4096
#define SEQ    2048
#define BATCH  2
#define NROWS  (BATCH * SEQ)   // 4096
#define HEADS  16
#define HDIM   64

// TDM: builtin exists on both toolchains (probe-verified); only the arity differs
// (ROCm7.2/clang-22 = 5 args, amdgpu-toolchain/clang-23 = 6 args). Gate on the
// device pass so the host pass (no amdgcn builtins) sees the fallback.
#if defined(__AMDGCN__) && __has_builtin(__builtin_amdgcn_tensor_load_to_lds)
#define USE_TDM 1
#else
#define USE_TDM 0
#endif

// ---------- helpers ----------

__device__ __forceinline__ bfT f2bf(float f) {
  unsigned u = __builtin_bit_cast(unsigned, f);
  u += 0x7FFFu + ((u >> 16) & 1u);            // round-to-nearest-even
  unsigned short hs = (unsigned short)(u >> 16);
  return __builtin_bit_cast(bfT, hs);
}

__device__ __forceinline__ v8f wmma_bf16(v16bf a, v16bf b, v8f c) {
  return __builtin_amdgcn_wmma_f32_16x16x32_bf16(
      /*neg_a=*/false, a, /*neg_b=*/false, b,
      /*c_mod=*/(short)0, c, /*reuse_a=*/false, /*reuse_b=*/false);
}

union AFrag { v16bf v; f32x4 f[2]; };

#if USE_TDM
// DMA a 128-row x 32-elem bf16 tile (row stride = rowStride elems) from global
// into LDS at byte offset lds_off, inserting 8 bf16 (4 dword) padding after each
// 64B row -> matches a bfT[128][40] C array exactly.
__device__ __forceinline__ void tdm_load_tile_128x32(unsigned lds_off,
                                                     const bfT* gsrc,
                                                     int rowStride) {
  unsigned long long ga = (unsigned long long)(uintptr_t)gsrc;
  u32x4 g0;
  g0[0] = 1u;                                        // count=1 (valid user D#)
  g0[1] = lds_off;                                   // LDS byte address
  g0[2] = (unsigned)ga;                              // global addr lo
  g0[3] = (unsigned)((ga >> 32) & 0x01FFFFFFull)     // global addr [56:32]
        | 0x80000000u;                               // type=2 ("image")
  i32x8 g1;
  g1[0] = (1 << 16)                                  // data_size = 2 bytes
        | (1 << 20)                                  // pad_enable
        | (3 << 22)                                  // pad_interval: 16 DW (64B)
        | (3 << 25);                                 // pad_amount: 4 DW (8 bf16)
  g1[1] = (32 << 16);                                // tensor_dim0 = 32 (tile-local)
  g1[2] = (128 << 16);                               // tensor_dim1 = 128
  g1[3] = (32 << 16);                                // tile_dim0 = 32
  g1[4] = 128;                                       // tile_dim1 = 128, tile_dim2 = 0
  g1[5] = rowStride;                                 // tensor_dim0_stride lo
  g1[6] = 0;
  g1[7] = 0;
  i32x4 g2 = (i32x4){0, 0, 0, 0};                    // 2D tile: groups 2/3 unused
  i32x4 g3 = (i32x4){0, 0, 0, 0};
#if __clang_major__ >= 23
  i32x8 gx = (i32x8){0, 0, 0, 0, 0, 0, 0, 0};
  __builtin_amdgcn_tensor_load_to_lds(g0, g1, g2, g3, gx, 0);
#else
  __builtin_amdgcn_tensor_load_to_lds(g0, g1, g2, g3, 0);
#endif
}
#endif

// ---------- fp32 -> bf16 conversion (plain) ----------

__global__ __launch_bounds__(256)
void cvt_bf16_kernel(const float* __restrict__ src, bfT* __restrict__ dst, int n) {
  int i = blockIdx.x * 256 + threadIdx.x;
  if (i < n) dst[i] = f2bf(src[i]);
}

// ---------- fp32 -> bf16 transposing conversion: W[K][N] -> Wt[N][K] ----------

__global__ __launch_bounds__(256)
void cvt_transpose_kernel(const float* __restrict__ W, bfT* __restrict__ Wt,
                          int K, int N) {
  __shared__ float tile[32][33];
  const int tx = threadIdx.x & 31;
  const int ty = threadIdx.x >> 5;                   // 0..7
  const int k0 = blockIdx.y * 32, n0 = blockIdx.x * 32;
#pragma unroll
  for (int i = 0; i < 4; i++) {
    const int r = ty * 4 + i;
    tile[r][tx] = W[(size_t)(k0 + r) * N + n0 + tx];
  }
  __syncthreads();
#pragma unroll
  for (int i = 0; i < 4; i++) {
    const int r = ty * 4 + i;
    Wt[(size_t)(n0 + r) * K + k0 + tx] = f2bf(tile[tx][r]);
  }
}

// ---------- bf16 GEMM: C(MxN) = A(MxK) * Bt(NxK)^T + bias ----------
// A row-major MxK, Bt row-major NxK (pre-transposed weight).
// 256 threads (8 waves), block tile 128x128, K-step 32, TDM double-buffered.
// wave grid 4(M) x 2(N); wave tile 32x64 -> 2x4 WMMA accumulators.

__global__ __launch_bounds__(256, 1)
void gemm_bf16_kernel(const bfT* __restrict__ A, const bfT* __restrict__ Bt,
                      const float* __restrict__ bias,
                      float* __restrict__ outF, bfT* __restrict__ outB,
                      bfT* __restrict__ outVt,
                      int M, int N, int K, int relu) {
  __shared__ __align__(16) bfT As[2][128][40];
  __shared__ __align__(16) bfT Bs[2][128][40];

  const int tid  = threadIdx.x;
  const int wid  = tid >> 5;
  const int lane = tid & 31;
  const int half = lane >> 4;
  const int l16  = lane & 15;
  const int wm   = wid & 3;
  const int wn   = wid >> 2;
  const int bm0  = blockIdx.y * 128;
  const int bn0  = blockIdx.x * 128;

  const bfT* Abase = A  + (size_t)bm0 * K;
  const bfT* Bbase = Bt + (size_t)bn0 * K;

  v8f acc[2][4];
  const v8f vzero = {0.f, 0.f, 0.f, 0.f, 0.f, 0.f, 0.f, 0.f};
  for (int i = 0; i < 2; i++)
    for (int j = 0; j < 4; j++) acc[i][j] = vzero;

#if USE_TDM
  const unsigned ldsA0 = (unsigned)(uintptr_t)&As[0][0][0];
  const unsigned ldsA1 = (unsigned)(uintptr_t)&As[1][0][0];
  const unsigned ldsB0 = (unsigned)(uintptr_t)&Bs[0][0][0];
  const unsigned ldsB1 = (unsigned)(uintptr_t)&Bs[1][0][0];
  if (wid == 0) {
    tdm_load_tile_128x32(ldsA0, Abase, K);
    tdm_load_tile_128x32(ldsB0, Bbase, K);
  }
#endif

  int buf = 0;
  for (int k0 = 0; k0 < K; k0 += 32) {
#if USE_TDM
    if (wid == 0) {
      if (k0 + 32 < K) {   // prefetch next K-slice into the other buffer
        tdm_load_tile_128x32(buf ? ldsA0 : ldsA1, Abase + k0 + 32, K);
        tdm_load_tile_128x32(buf ? ldsB0 : ldsB1, Bbase + k0 + 32, K);
        __builtin_amdgcn_s_wait_tensorcnt(2);   // current pair retired (in-order)
      } else {
        __builtin_amdgcn_s_wait_tensorcnt(0);
      }
    }
    __syncthreads();
    const int rb = buf;
#else
    for (int c = tid; c < 512; c += 256) {
      const int row = c >> 2, col8 = (c & 3) << 3;
      *(f32x4*)&As[0][row][col8] = *(const f32x4*)(Abase + (size_t)row * K + k0 + col8);
      *(f32x4*)&Bs[0][row][col8] = *(const f32x4*)(Bbase + (size_t)row * K + k0 + col8);
    }
    __syncthreads();
    const int rb = 0;
#endif

    // B fragments: b[e<8] = Bt[col][half*8+e], b[8+e] = Bt[col][half*8+16+e]
    v16bf bfrag[4];
#pragma unroll
    for (int j = 0; j < 4; j++) {
      const bfT* bp = &Bs[rb][wn * 64 + j * 16 + l16][half * 8];
      AFrag t;
      t.f[0] = *(const f32x4*)bp;
      t.f[1] = *(const f32x4*)(bp + 16);
      bfrag[j] = t.v;
    }
    // A fragments: a[e<8] = A[row][half*8+e], a[8+e] = A[row][half*8+16+e]
#pragma unroll
    for (int i = 0; i < 2; i++) {
      const bfT* ap = &As[rb][wm * 32 + i * 16 + l16][half * 8];
      AFrag a;
      a.f[0] = *(const f32x4*)ap;
      a.f[1] = *(const f32x4*)(ap + 16);
#pragma unroll
      for (int j = 0; j < 4; j++)
        acc[i][j] = wmma_bf16(a.v, bfrag[j], acc[i][j]);
    }
    __syncthreads();
    buf ^= 1;
  }

  // epilogue: C layout -> lane l16 = N col; VGPR r -> M row = r + 8*half
  if (outVt) {
    // V projection: write [B][H][Hd][S] head-transposed
    for (int i = 0; i < 2; i++)
      for (int j = 0; j < 4; j++) {
        const int gm = bm0 + wm * 32 + i * 16 + half * 8;
        const int gn = bn0 + wn * 64 + j * 16 + l16;
        const float bb = bias ? bias[gn] : 0.0f;
        const int h = gn >> 6, hd = gn & (HDIM - 1);
#pragma unroll
        for (int r = 0; r < 8; r++) {
          const float v = acc[i][j][r] + bb;
          const int row = gm + r;
          const int bi = row >> 11, s = row & (SEQ - 1);
          outVt[(((size_t)bi * HEADS + h) * HDIM + hd) * SEQ + s] = f2bf(v);
        }
      }
  } else {
    for (int i = 0; i < 2; i++)
      for (int j = 0; j < 4; j++) {
        const int gm = bm0 + wm * 32 + i * 16 + half * 8;
        const int gn = bn0 + wn * 64 + j * 16 + l16;
        const float bb = bias ? bias[gn] : 0.0f;
#pragma unroll
        for (int r = 0; r < 8; r++) {
          float v = acc[i][j][r] + bb;
          if (relu) v = fmaxf(v, 0.0f);
          const size_t idx = (size_t)(gm + r) * N + gn;
          if (outF) outF[idx] = v;
          if (outB) outB[idx] = f2bf(v);
        }
      }
  }
}

// ---------- causal flash attention ----------
// Q,K in [B,S,D] bf16 (head h = cols h*64..); V in [B][H][Hd][S] bf16.
// block = 128 threads (4 waves); each wave owns 16 q-rows; grid (S/64, B*H)

__global__ __launch_bounds__(128, 1)
void attn_fwd_kernel(const bfT* __restrict__ Q, const bfT* __restrict__ Km,
                     const bfT* __restrict__ Vt, bfT* __restrict__ O) {
  __shared__ __align__(16) bfT Pt[4][16][64];   // per-wave P tile (bf16)

  const int tid  = threadIdx.x;
  const int wid  = tid >> 5;
  const int lane = tid & 31;
  const int half = lane >> 4;
  const int l16  = lane & 15;
  const int bh = blockIdx.y, b = bh >> 4, h = bh & 15;
  const int q0 = blockIdx.x * 64 + wid * 16;
  const size_t base  = ((size_t)b * SEQ) * HIDDEN + (size_t)h * HDIM;
  const bfT* vb = Vt + (size_t)bh * HDIM * SEQ;

  // Q A-fragments (held in registers for the whole scan)
  const bfT* qp = Q + base + (size_t)(q0 + l16) * HIDDEN + half * 8;
  AFrag aq0, aq1;
  aq0.f[0] = *(const f32x4*)qp;        aq0.f[1] = *(const f32x4*)(qp + 16);
  aq1.f[0] = *(const f32x4*)(qp + 32); aq1.f[1] = *(const f32x4*)(qp + 48);

  const v8f vzero = {0.f, 0.f, 0.f, 0.f, 0.f, 0.f, 0.f, 0.f};
  v8f Ofr[4];
  for (int t = 0; t < 4; t++) Ofr[t] = vzero;
  float mrow[8], lrow[8];
#pragma unroll
  for (int r = 0; r < 8; r++) { mrow[r] = -1e30f; lrow[r] = 0.0f; }

  const int kcmax = q0 >> 6;
  for (int kc = 0; kc <= kcmax; kc++) {
    const int kv0 = kc * 64;

    // prefetch next chunk's K rows / V columns into cache (global_prefetch_b8)
    if (kc < kcmax) {
      const bfT* nk = Km + base + (size_t)(kv0 + 64 + lane) * HIDDEN;
      __builtin_prefetch(nk, 0, 1);
      __builtin_prefetch(nk + (size_t)32 * HIDDEN, 0, 1);
      const bfT* nv = vb + (size_t)lane * SEQ + kv0 + 64;
      __builtin_prefetch(nv, 0, 1);
      __builtin_prefetch(nv + (size_t)32 * SEQ, 0, 1);
    }

    // ---- S = Q K^T / sqrt(Hd), causal mask ----
    v8f sc[4];
#pragma unroll
    for (int ns = 0; ns < 4; ns++) {
      const int col = kv0 + ns * 16 + l16;
      const bfT* kp = Km + base + (size_t)col * HIDDEN + half * 8;
      AFrag bk0, bk1;
      bk0.f[0] = *(const f32x4*)kp;        bk0.f[1] = *(const f32x4*)(kp + 16);
      bk1.f[0] = *(const f32x4*)(kp + 32); bk1.f[1] = *(const f32x4*)(kp + 48);
      v8f c = vzero;
      c = wmma_bf16(aq0.v, bk0.v, c);
      c = wmma_bf16(aq1.v, bk1.v, c);
#pragma unroll
      for (int r = 0; r < 8; r++) {
        float s = c[r] * 0.125f;                    // 1/sqrt(64)
        if (col > q0 + half * 8 + r) s = -1e30f;    // causal mask
        c[r] = s;
      }
      sc[ns] = c;
    }

    // ---- online softmax ----
    float fac[8], mnew[8];
#pragma unroll
    for (int r = 0; r < 8; r++) {
      float rm = fmaxf(fmaxf(sc[0][r], sc[1][r]), fmaxf(sc[2][r], sc[3][r]));
      rm = fmaxf(rm, __shfl_xor(rm, 1));
      rm = fmaxf(rm, __shfl_xor(rm, 2));
      rm = fmaxf(rm, __shfl_xor(rm, 4));
      rm = fmaxf(rm, __shfl_xor(rm, 8));
      const float mn = fmaxf(mrow[r], rm);
      fac[r] = __expf(mrow[r] - mn);
      mrow[r] = mn; mnew[r] = mn;
    }
#pragma unroll
    for (int r = 0; r < 8; r++) {
      float ps = 0.0f;
#pragma unroll
      for (int ns = 0; ns < 4; ns++) {
        const float pp = __expf(sc[ns][r] - mnew[r]);
        ps += pp;
        Pt[wid][half * 8 + r][ns * 16 + l16] = f2bf(pp);
      }
      ps += __shfl_xor(ps, 1);
      ps += __shfl_xor(ps, 2);
      ps += __shfl_xor(ps, 4);
      ps += __shfl_xor(ps, 8);
      lrow[r] = lrow[r] * fac[r] + ps;
#pragma unroll
      for (int t = 0; t < 4; t++) Ofr[t][r] *= fac[r];
    }

    __asm__ volatile("s_wait_dscnt 0" ::: "memory");  // wave-private LDS RAW

    // ---- O += P V ----
    const bfT* pp = &Pt[wid][l16][half * 8];
    AFrag ap0, ap1;
    ap0.f[0] = *(const f32x4*)pp;        ap0.f[1] = *(const f32x4*)(pp + 16);
    ap1.f[0] = *(const f32x4*)(pp + 32); ap1.f[1] = *(const f32x4*)(pp + 48);
#pragma unroll
    for (int t = 0; t < 4; t++) {
      // B frag f: b[e<8] = V[kv0 + f*32 + half*8 + e][t*16+l16], b[8+e] = V[kv0 + f*32 + half*8 + 16 + e][..] (Vt row-contig)
      const bfT* vp0 = vb + (size_t)(t * 16 + l16) * SEQ + kv0 + half * 8;
      const bfT* vp1 = vp0 + 32;
      AFrag bv0, bv1;
      bv0.f[0] = *(const f32x4*)vp0;       bv0.f[1] = *(const f32x4*)(vp0 + 16);
      bv1.f[0] = *(const f32x4*)vp1;       bv1.f[1] = *(const f32x4*)(vp1 + 16);
      Ofr[t] = wmma_bf16(ap0.v, bv0.v, Ofr[t]);
      Ofr[t] = wmma_bf16(ap1.v, bv1.v, Ofr[t]);
    }
    __asm__ volatile("s_wait_dscnt 0" ::: "memory");  // WAR fence for next chunk
  }

  // ---- normalize and store (layout [B,S,H,Hd] == [B,S,D]) ----
#pragma unroll
  for (int t = 0; t < 4; t++) {
#pragma unroll
    for (int r = 0; r < 8; r++) {
      const float o = Ofr[t][r] / lrow[r];
      const size_t idx = base + (size_t)(q0 + half * 8 + r) * HIDDEN + t * 16 + l16;
      O[idx] = f2bf(o);
    }
  }
}

// ---------- fused residual-add + LayerNorm ----------

__global__ __launch_bounds__(256)
void add_ln_kernel(const float* __restrict__ x, const float* __restrict__ resid,
                   const float* __restrict__ g, const float* __restrict__ be,
                   float* __restrict__ outF, bfT* __restrict__ outB) {
  __shared__ float sS[8], sQ[8];
  const int row = blockIdx.x;
  const int tid = threadIdx.x;
  const int wid = tid >> 5, lane = tid & 31;
  const size_t b0 = (size_t)row * HIDDEN;

  float v[4]; float s = 0.f, q = 0.f;
#pragma unroll
  for (int i = 0; i < 4; i++) {
    const int c = i * 256 + tid;
    v[i] = x[b0 + c] + resid[b0 + c];
    s += v[i]; q += v[i] * v[i];
  }
#pragma unroll
  for (int m = 1; m < 32; m <<= 1) { s += __shfl_xor(s, m); q += __shfl_xor(q, m); }
  if (lane == 0) { sS[wid] = s; sQ[wid] = q; }
  __syncthreads();
  float ts = 0.f, tq = 0.f;
#pragma unroll
  for (int k = 0; k < 8; k++) { ts += sS[k]; tq += sQ[k]; }
  const float mu  = ts * (1.0f / HIDDEN);
  const float var = tq * (1.0f / HIDDEN) - mu * mu;
  const float rs  = rsqrtf(var + 1e-6f);
#pragma unroll
  for (int i = 0; i < 4; i++) {
    const int c = i * 256 + tid;
    const float o = (v[i] - mu) * rs * g[c] + be[c];
    outF[b0 + c] = o;
    if (outB) outB[b0 + c] = f2bf(o);
  }
}

// ---------- host orchestration ----------

extern "C" void kernel_launch(void* const* d_in, const int* in_sizes, int n_in,
                              void* d_out, int out_size, void* d_ws, size_t ws_size,
                              hipStream_t stream) {
  (void)in_sizes; (void)n_in; (void)out_size; (void)ws_size;
  const float* x   = (const float*)d_in[0];
  const float* Wq  = (const float*)d_in[1];
  const float* bq  = (const float*)d_in[2];
  const float* Wk  = (const float*)d_in[3];
  const float* bk  = (const float*)d_in[4];
  const float* Wv  = (const float*)d_in[5];
  const float* bv  = (const float*)d_in[6];
  const float* Wo  = (const float*)d_in[7];
  const float* bo  = (const float*)d_in[8];
  const float* g1  = (const float*)d_in[9];
  const float* b1  = (const float*)d_in[10];
  const float* Wup = (const float*)d_in[11];
  const float* bup = (const float*)d_in[12];
  const float* Wdn = (const float*)d_in[13];
  const float* bdn = (const float*)d_in[14];
  const float* g2  = (const float*)d_in[15];
  const float* b2  = (const float*)d_in[16];

  const int M = NROWS, D = HIDDEN, F = FFDIM;
  char* p = (char*)d_ws;
  auto allocB = [&](size_t n) { bfT* r = (bfT*)p; p += ((n * sizeof(bfT)  + 255) & ~(size_t)255); return r; };
  auto allocF = [&](size_t n) { float* r = (float*)p; p += ((n * sizeof(float) + 255) & ~(size_t)255); return r; };

  bfT*   xb   = allocB((size_t)M * D);   // x in bf16 (row-major)
  bfT*   Wb   = allocB((size_t)D * F);   // shared transposed-weight buffer
  bfT*   Qb   = allocB((size_t)M * D);
  bfT*   Kb   = allocB((size_t)M * D);
  bfT*   Vt   = allocB((size_t)M * D);   // V in [B][H][Hd][S]
  bfT*   midb = allocB((size_t)M * D);   // attention output (bf16)
  float* tF   = allocF((size_t)M * D);   // attn_out fp32, later FFN-down fp32
  float* x1f  = allocF((size_t)M * D);   // after LN1 (fp32 residual)
  bfT*   x1b  = allocB((size_t)M * D);   // after LN1 (bf16 for FFN GEMM)
  bfT*   upb  = allocB((size_t)M * F);   // relu(x1 Wup + bup) in bf16

  const dim3 blk256(256);
  auto cvt = [&](const float* s, bfT* d, size_t n) {
    cvt_bf16_kernel<<<dim3((unsigned)((n + 255) / 256)), blk256, 0, stream>>>(s, d, (int)n);
  };
  auto cvtT = [&](const float* w, bfT* wt, int K, int N) {   // W[K][N] -> Wt[N][K]
    cvt_transpose_kernel<<<dim3(N / 32, K / 32), blk256, 0, stream>>>(w, wt, K, N);
  };

  cvt(x, xb, (size_t)M * D);

  // Q, K, V projections (transposed weight in shared buffer; stream order serializes reuse)
  cvtT(Wq, Wb, D, D);
  gemm_bf16_kernel<<<dim3(D / 128, M / 128), blk256, 0, stream>>>(xb, Wb, bq, nullptr, Qb, nullptr, M, D, D, 0);
  cvtT(Wk, Wb, D, D);
  gemm_bf16_kernel<<<dim3(D / 128, M / 128), blk256, 0, stream>>>(xb, Wb, bk, nullptr, Kb, nullptr, M, D, D, 0);
  cvtT(Wv, Wb, D, D);
  gemm_bf16_kernel<<<dim3(D / 128, M / 128), blk256, 0, stream>>>(xb, Wb, bv, nullptr, nullptr, Vt, M, D, D, 0);

  // causal attention
  attn_fwd_kernel<<<dim3(SEQ / 64, BATCH * HEADS), dim3(128), 0, stream>>>(Qb, Kb, Vt, midb);

  // output projection (fp32 out for residual)
  cvtT(Wo, Wb, D, D);
  gemm_bf16_kernel<<<dim3(D / 128, M / 128), blk256, 0, stream>>>(midb, Wb, bo, tF, nullptr, nullptr, M, D, D, 0);

  // LN1: x + attn_out
  add_ln_kernel<<<dim3(M), blk256, 0, stream>>>(x, tF, g1, b1, x1f, x1b);

  // FFN up (ReLU fused, bf16 out)
  cvtT(Wup, Wb, D, F);
  gemm_bf16_kernel<<<dim3(F / 128, M / 128), blk256, 0, stream>>>(x1b, Wb, bup, nullptr, upb, nullptr, M, F, D, 1);

  // FFN down (fp32 out)
  cvtT(Wdn, Wb, F, D);
  gemm_bf16_kernel<<<dim3(D / 128, M / 128), blk256, 0, stream>>>(upb, Wb, bdn, tF, nullptr, nullptr, M, D, F, 0);

  // LN2 -> final output (fp32)
  add_ln_kernel<<<dim3(M), blk256, 0, stream>>>(x1f, tF, g2, b2, (float*)d_out, nullptr);
}
